// Attention_558345749040
// MI455X (gfx1250) — matrix operand order repaired and checked by reading, not compiled, not run on hardware
//
#include <hip/hip_runtime.h>

// ---------------------------------------------------------------------------
// Fused "attention" (elementwise variant) for MI455X / gfx1250, wave32.
//   out[b,s,h*64+j] = softmax_j( Qp*Kp/8 - (j>=V_len)*1e12 ) * Vp * (s<Q_len)
// Qp/Kp/Vp are the three 512x512 projections done with split-bf16 (hi/lo)
// V_WMMA_F32_16X16X32_BF16 (3-term Markidis) for near-fp32 accuracy.
// MT=128 row tile, per-wave 2x2 WMMA tiles, float4 global loads, software-
// pipelined global->LDS staging.
// ---------------------------------------------------------------------------

typedef __bf16 bf16_t;
typedef __attribute__((ext_vector_type(16))) __bf16 v16bf;
typedef __attribute__((ext_vector_type(8)))  __bf16 v8bf;
typedef __attribute__((ext_vector_type(4)))  __bf16 v4bf;
typedef __attribute__((ext_vector_type(8)))  float  v8f;
typedef __attribute__((ext_vector_type(4)))  float  v4f;

#define BB 8
#define SS 4096
#define EE 512
#define HH 8
#define DD 64
#define MT 128                // row tile per workgroup
#define KC 32                 // K chunk (matches 16x16x32 WMMA depth)
#define KSTEPS (EE / KC)      // 16
#define ASTR 40               // bf16 elems per staged row (32 + 8 pad, 16B-mult)
#define RSTR 66               // f32 elems per result row (64 + 2 pad)

#define WMMA_BF16(A, B, C) \
  __builtin_amdgcn_wmma_f32_16x16x32_bf16(false, (A), false, (B), (short)0, (C), false, false)

// A fragment: lane holds row (lane&15); 8 K at off and 8 K at off+16.
__device__ __forceinline__ v16bf frag_a(const bf16_t* row, int off) {
  union { v16bf v; v8bf h[2]; } u;
  u.h[0] = *(const v8bf*)(row + off);
  u.h[1] = *(const v8bf*)(row + off + 16);
  return u.v;
}
// B fragment: lane holds column (lane&15); 8 K at off and 8 K at off+16.
__device__ __forceinline__ v16bf frag_b(const bf16_t* row, int off) {
  union { v16bf v; v8bf h[2]; } u;
  u.h[0] = *(const v8bf*)(row + off);
  u.h[1] = *(const v8bf*)(row + off + 16);
  return u.v;
}

__global__ __launch_bounds__(256) void fused_mha_kernel(
    const float* __restrict__ Qseq, const float* __restrict__ Kseq,
    const float* __restrict__ Vseq, const int* __restrict__ Qlen,
    const int* __restrict__ Vlen, const float* __restrict__ WQ,
    const float* __restrict__ WK, const float* __restrict__ WV,
    float* __restrict__ out)
{
  __shared__ bf16_t AsH[MT][ASTR];   // input tile, bf16 hi   (10 KB)
  __shared__ bf16_t AsL[MT][ASTR];   // input tile, bf16 lo   (10 KB)
  __shared__ bf16_t BsH[DD][ASTR];   // weights ([n][k] transposed), hi (5 KB)
  __shared__ bf16_t BsL[DD][ASTR];   // weights lo                      (5 KB)
  __shared__ float  QKs[MT][RSTR];   // Qp, then fused Qp*Kp/8 (33 KB)
  __shared__ float  Vs [MT][RSTR];   // Vp                     (33 KB)

  const int tid   = threadIdx.x;
  const int lane  = tid & 31;
  const int wave  = tid >> 5;
  const int tileS = blockIdx.x;   // 0..31
  const int head  = blockIdx.y;   // 0..7
  const int b     = blockIdx.z;   // 0..7
  const int row0  = tileS * MT;   // row offset within S

  const float* Xbase[3] = {
      Qseq + ((size_t)b * SS + row0) * EE,
      Kseq + ((size_t)b * SS + row0) * EE,
      Vseq + ((size_t)b * SS + row0) * EE };
  const float* Wbase[3] = { WQ + head * DD, WK + head * DD, WV + head * DD };

  // Wave -> 32x32 output slab: 2x2 WMMA tiles.
  const int wr    = wave & 3;           // row group 0..3 (rows wr*32..+31)
  const int wc    = wave >> 2;          // col group 0..1 (cols wc*32..+31)
  const int l15   = lane & 15;
  const int hi16  = lane >> 4;
  const int aoff  = hi16 * 8;           // A K-chunk select
  const int boff  = hi16 * 8;           // B K-chunk select
  const int arow0 = wr * 32 + l15;      // A row, first tile
  const int brow0 = wc * 32 + l15;      // B column, first tile
  const int crow  = hi16 * 8;           // C row base within tile (+r)
  const int col0  = wc * 32 + l15;      // C column, first tile

#pragma unroll
  for (int p = 0; p < 3; ++p) {
    const float* X = Xbase[p];
    const float* W = Wbase[p];
    v8f acc00 = {0.f,0.f,0.f,0.f,0.f,0.f,0.f,0.f};
    v8f acc01 = acc00, acc10 = acc00, acc11 = acc00;

    // Software pipeline: next global chunk held in registers (float4 loads).
    v4f a4[4], b4[2];
#pragma unroll
    for (int i = 0; i < 4; ++i) {
      int f = tid + (i << 8);                       // A: 128x32 = 1024 float4
      a4[i] = *(const v4f*)(X + (size_t)(f >> 3) * EE + (f & 7) * 4);
    }
#pragma unroll
    for (int i = 0; i < 2; ++i) {
      int f = tid + (i << 8);                       // B: 32x64 = 512 float4
      b4[i] = *(const v4f*)(W + (size_t)(f >> 4) * EE + (f & 15) * 4);
    }

    for (int kb = 0; kb < KSTEPS; ++kb) {
      __syncthreads();  // previous chunk's LDS fully consumed
      // Convert f32 -> bf16 hi/lo and stage (B stored transposed [n][k]).
#pragma unroll
      for (int i = 0; i < 4; ++i) {
        int f = tid + (i << 8);
        int m = f >> 3, k0 = (f & 7) * 4;
        v4bf h4, l4;
#pragma unroll
        for (int j = 0; j < 4; ++j) {
          float x = a4[i][j];
          bf16_t h = (bf16_t)x;
          h4[j] = h;
          l4[j] = (bf16_t)(x - (float)h);
        }
        *(v4bf*)&AsH[m][k0] = h4;
        *(v4bf*)&AsL[m][k0] = l4;
      }
#pragma unroll
      for (int i = 0; i < 2; ++i) {
        int f = tid + (i << 8);
        int k = f >> 4, n0 = (f & 15) * 4;
#pragma unroll
        for (int j = 0; j < 4; ++j) {
          float x = b4[i][j];
          bf16_t h = (bf16_t)x;
          BsH[n0 + j][k] = h;
          BsL[n0 + j][k] = (bf16_t)(x - (float)h);
        }
      }
      __syncthreads();
      // Issue next chunk's global loads; they complete under the WMMAs.
      if (kb + 1 < KSTEPS) {
        const float* Xk = X + (size_t)(kb + 1) * KC;
        const float* Wk = W + (size_t)(kb + 1) * KC * EE;
#pragma unroll
        for (int i = 0; i < 4; ++i) {
          int f = tid + (i << 8);
          a4[i] = *(const v4f*)(Xk + (size_t)(f >> 3) * EE + (f & 7) * 4);
        }
#pragma unroll
        for (int i = 0; i < 2; ++i) {
          int f = tid + (i << 8);
          b4[i] = *(const v4f*)(Wk + (size_t)(f >> 4) * EE + (f & 15) * 4);
        }
      }

      v16bf a0h = frag_a(&AsH[arow0][0],      aoff);
      v16bf a0l = frag_a(&AsL[arow0][0],      aoff);
      v16bf a1h = frag_a(&AsH[arow0 + 16][0], aoff);
      v16bf a1l = frag_a(&AsL[arow0 + 16][0], aoff);
      v16bf b0h = frag_b(&BsH[brow0][0],      boff);
      v16bf b0l = frag_b(&BsL[brow0][0],      boff);
      v16bf b1h = frag_b(&BsH[brow0 + 16][0], boff);
      v16bf b1l = frag_b(&BsL[brow0 + 16][0], boff);

      // 3-term split-bf16 product, fp32 accumulate; 12 independent-chain WMMAs.
      acc00 = WMMA_BF16(a0l, b0h, acc00);
      acc01 = WMMA_BF16(a0l, b1h, acc01);
      acc10 = WMMA_BF16(a1l, b0h, acc10);
      acc11 = WMMA_BF16(a1l, b1h, acc11);
      acc00 = WMMA_BF16(a0h, b0l, acc00);
      acc01 = WMMA_BF16(a0h, b1l, acc01);
      acc10 = WMMA_BF16(a1h, b0l, acc10);
      acc11 = WMMA_BF16(a1h, b1l, acc11);
      acc00 = WMMA_BF16(a0h, b0h, acc00);
      acc01 = WMMA_BF16(a0h, b1h, acc01);
      acc10 = WMMA_BF16(a1h, b0h, acc10);
      acc11 = WMMA_BF16(a1h, b1h, acc11);
    }

    // Epilogue per projection. C layout: VGPR r -> row r+8*(lane>>4).
#pragma unroll
    for (int r = 0; r < 8; ++r) {
      int rr0 = wr * 32 + crow + r;
      int rr1 = rr0 + 16;
      int cc0 = col0, cc1 = col0 + 16;
      if (p == 0) {            // Qp
        QKs[rr0][cc0] = acc00[r];
        QKs[rr0][cc1] = acc01[r];
        QKs[rr1][cc0] = acc10[r];
        QKs[rr1][cc1] = acc11[r];
      } else if (p == 1) {     // fuse A = Qp*Kp/sqrt(64), in place (same thread)
        QKs[rr0][cc0] = QKs[rr0][cc0] * acc00[r] * 0.125f;
        QKs[rr0][cc1] = QKs[rr0][cc1] * acc01[r] * 0.125f;
        QKs[rr1][cc0] = QKs[rr1][cc0] * acc10[r] * 0.125f;
        QKs[rr1][cc1] = QKs[rr1][cc1] * acc11[r] * 0.125f;
      } else {                 // Vp
        Vs[rr0][cc0] = acc00[r];
        Vs[rr0][cc1] = acc01[r];
        Vs[rr1][cc0] = acc10[r];
        Vs[rr1][cc1] = acc11[r];
      }
    }
  }

  __syncthreads();

  // Per-row masked softmax over d=64, times Vp, times Q_len row mask.
  if (tid < MT) {
    const int srow = row0 + tid;
    const int vl = Vlen[b];
    const int ql = Qlen[b];
    float m = -3.0e38f;
    for (int j = 0; j < DD; ++j) {
      float aj = QKs[tid][j] - ((j >= vl) ? 1.0e12f : 0.0f);
      m = fmaxf(m, aj);
    }
    float sum = 0.0f;
    for (int j = 0; j < DD; ++j) {
      float aj = QKs[tid][j] - ((j >= vl) ? 1.0e12f : 0.0f);
      sum += __expf(aj - m);
    }
    const float inv = 1.0f / sum;
    const float qm  = (srow < ql) ? 1.0f : 0.0f;
    float* o = out + ((size_t)b * SS + srow) * (HH * DD) + head * DD;
    for (int j = 0; j < DD; ++j) {
      float aj = QKs[tid][j] - ((j >= vl) ? 1.0e12f : 0.0f);
      o[j] = __expf(aj - m) * inv * Vs[tid][j] * qm;
    }
  }
}

extern "C" void kernel_launch(void* const* d_in, const int* in_sizes, int n_in,
                              void* d_out, int out_size, void* d_ws, size_t ws_size,
                              hipStream_t stream) {
  (void)in_sizes; (void)n_in; (void)out_size; (void)d_ws; (void)ws_size;
  const float* Qseq = (const float*)d_in[0];
  const float* Kseq = (const float*)d_in[1];
  const float* Vseq = (const float*)d_in[2];
  const int*   Qlen = (const int*)d_in[3];
  const int*   Vlen = (const int*)d_in[4];
  const float* WQ   = (const float*)d_in[5];
  const float* WK   = (const float*)d_in[6];
  const float* WV   = (const float*)d_in[7];
  float* out = (float*)d_out;

  dim3 grid(SS / MT, HH, BB);   // 32 x 8 x 8 = 2048 workgroups
  dim3 block(256);              // 8 waves (wave32)
  hipLaunchKernelGGL(fused_mha_kernel, grid, block, 0, stream,
                     Qseq, Kseq, Vseq, Qlen, Vlen, WQ, WK, WV, out);
}
